// WindowedAttention_46892452937868
// MI455X (gfx1250) — matrix operand order repaired and checked by reading, not compiled, not run on hardware
//
#include <hip/hip_runtime.h>

typedef __attribute__((ext_vector_type(16))) _Float16 v16h;
typedef __attribute__((ext_vector_type(8)))  float    v8f;

#define NWIN  200
#define NPOS  224       // window tokens padded 196 -> 224 (14 WMMA tiles)
#define HEADS 16
#define DHEAD 64
#define CH    1024
#define WROWS 39200     // 200 windows * 196 tokens
#define XROWS 32768     // 8 * 64 * 64

union HV { uint4 u[2]; _Float16 h[16]; v16h v; };

__device__ __forceinline__ v8f zero8() {
  v8f z = {0.f,0.f,0.f,0.f,0.f,0.f,0.f,0.f};
  return z;
}

// ---- CDNA5 async global->LDS copy (ASYNCcnt tracked), 16 bytes per op ----
// Generic pointers to LDS carry the LDS byte offset in their low 32 bits
// (ISA 10.2 aperture mapping), so the low dword is the vdst LDS address.
__device__ __forceinline__ void async_b128(const void* g, void* l) {
  unsigned lds_addr = (unsigned)(size_t)l;
  asm volatile("global_load_async_to_lds_b128 %0, %1, off"
               :: "v"(lds_addr), "v"(g) : "memory");
}
__device__ __forceinline__ void wait_async0() {
  asm volatile("s_wait_asynccnt 0x0" ::: "memory");
}

// A operand (16x32 f16): lane holds row M = lane%16, half h -> K = h + (h/8)*8 + (lane/16)*8
__device__ __forceinline__ v16h ld_a(const _Float16* base, int ld) {
  const int lane = threadIdx.x & 31;
  const _Float16* p = base + (lane & 15) * ld + ((lane >> 4) << 3);
  HV r;
  r.u[0] = *(const uint4*)(p);
  r.u[1] = *(const uint4*)(p + 16);
  return r.v;
}

// B operand (32x16 f16): lane holds col N = lane%16, half h -> K = h + (h/8)*8 + (lane/16)*8
__device__ __forceinline__ v16h ld_b(const _Float16* base, int ld) {
  const int lane = threadIdx.x & 31;
  const _Float16* p = base + (lane & 15) * ld + ((lane >> 4) << 3);
  HV r;
  r.u[0] = *(const uint4*)(p);
  r.u[1] = *(const uint4*)(p + 16);
  return r.v;
}

__device__ __forceinline__ v8f wmma_f16(v16h a, v16h b, v8f c) {
  return __builtin_amdgcn_wmma_f32_16x16x32_f16(false, a, false, b, (short)0, c, false, false);
}

// windowed row (0..39199) -> source row in x (B*4096) or -1 if padding
__device__ __forceinline__ int window_row_to_src(int wr) {
  int w = wr / 196, pos = wr % 196;
  int b = w / 25, rem = w % 25;
  int wi = rem / 5, wj = rem % 5;
  int ph = wi * 14 + pos / 14;
  int pw = wj * 14 + pos % 14;
  if (ph >= 64 || pw >= 64) return -1;
  return b * 4096 + ph * 64 + pw;
}

// fp32 [K][N] -> f16 transposed [N][K] (so GEMM B tiles are contiguous in K)
__global__ void cvt_transpose_f16(const float* __restrict__ src, _Float16* __restrict__ dst,
                                  int K, int N) {
  int i = blockIdx.x * blockDim.x + threadIdx.x;
  if (i < K * N) {
    int k = i / N, n = i % N;
    dst[(size_t)n * K + k] = (_Float16)src[i];
  }
}

// ---------------- Tiled WMMA GEMM: C[M,N] = A[M,K] * Bt[N,K]^T + bias ----------------
// Block 256 threads = 8 waves (4 along M, 2 along N). Tile 128x128, K-step 32.
// Bt is the pre-transposed weight matrix [N][K]; both A (f16 path) and B tiles are
// staged via global_load_async_to_lds_b128.
template<bool GATHER, bool OUTF16>
__global__ __launch_bounds__(256) void gemm_wmma(
    const float* __restrict__ Af, const _Float16* __restrict__ Ah,
    const _Float16* __restrict__ Bt, const float* __restrict__ bias,
    _Float16* __restrict__ out_h, float* __restrict__ out_f,
    int Mrows, int N, int K)
{
  __shared__ alignas(16) _Float16 As[128 * 40];   // [m][k], pad 40
  __shared__ alignas(16) _Float16 Bs[128 * 40];   // [n][k], pad 40

  const int t = threadIdx.x;
  const int lane = t & 31, wid = t >> 5;
  const int n_lane = lane & 15, hi = lane >> 4;
  const int wm = wid & 3, wn = wid >> 2;
  const long bm = (long)blockIdx.x * 128;
  const long bn = (long)blockIdx.y * 128;

  v8f acc[2][4];
#pragma unroll
  for (int i = 0; i < 2; ++i)
#pragma unroll
    for (int j = 0; j < 4; ++j) acc[i][j] = zero8();

  const int arow = t >> 1;              // 0..127 (also B tile row n)
  const int acol = (t & 1) << 4;        // 0 or 16 (also B tile k chunk)
  const long gRow = bm + arow;
  long srcRow;
  if (GATHER)
    srcRow = (gRow < Mrows) ? (long)window_row_to_src((int)gRow) : -1;
  else
    srcRow = (gRow < Mrows) ? gRow : 0;   // non-gather callers have M % 128 == 0

  for (int kt = 0; kt < K; kt += 32) {
    __syncthreads();
    // ---- stage A tile (128 x 32) ----
    if (GATHER) {
      union { float4 v[4]; float f[16]; } buf;
      if (srcRow >= 0) {
        const float4* ap = (const float4*)(Af + srcRow * (long)K + kt + acol);
        buf.v[0] = ap[0]; buf.v[1] = ap[1]; buf.v[2] = ap[2]; buf.v[3] = ap[3];
      } else {
#pragma unroll
        for (int i = 0; i < 16; ++i) buf.f[i] = 0.f;
      }
      _Float16* d = As + arow * 40 + acol;
#pragma unroll
      for (int i = 0; i < 16; ++i) d[i] = (_Float16)buf.f[i];
    } else {
      const _Float16* g = Ah + srcRow * (long)K + kt + acol;
      _Float16* l = As + arow * 40 + acol;
      async_b128(g,     l);
      async_b128(g + 8, l + 8);
    }
    // ---- stage B tile (128 rows of Bt x 32 k) via async copy ----
    {
      const _Float16* g = Bt + (size_t)(bn + arow) * K + kt + acol;
      _Float16* l = Bs + arow * 40 + acol;
      async_b128(g,     l);
      async_b128(g + 8, l + 8);
      if (kt + 32 < K)
        __builtin_prefetch(g + 32, 0, 3);
    }
    wait_async0();
    __syncthreads();
    // ---- compute ----
    v16h a0 = ld_a(As + (wm * 32)      * 40, 40);
    v16h a1 = ld_a(As + (wm * 32 + 16) * 40, 40);
#pragma unroll
    for (int nt = 0; nt < 4; ++nt) {
      v16h b = ld_b(Bs + (wn * 64 + nt * 16) * 40, 40);
      acc[0][nt] = wmma_f16(a0, b, acc[0][nt]);
      acc[1][nt] = wmma_f16(a1, b, acc[1][nt]);
    }
  }

  // ---- epilogue: D layout M = r + 8*hi, N = n_lane ----
#pragma unroll
  for (int mt = 0; mt < 2; ++mt) {
#pragma unroll
    for (int r = 0; r < 8; ++r) {
      long row = bm + wm * 32 + mt * 16 + r + 8 * hi;
      if (row < Mrows) {
#pragma unroll
        for (int nt = 0; nt < 4; ++nt) {
          long col = bn + wn * 64 + nt * 16 + n_lane;
          float v = acc[mt][nt][r] + bias[col];
          if (OUTF16) out_h[row * N + col] = (_Float16)v;
          else        out_f[row * N + col] = v;
        }
      }
    }
  }
}

// ---------------- LayerNorm(q,k) + scatter to attention layouts ----------------
// One block per (window, padded pos). q_h/k_h: [w][head][224][64]; v_t: [w][head][64][224].
__global__ __launch_bounds__(256) void ln_scatter(
    const _Float16* __restrict__ qkv_raw,
    const float* __restrict__ qn_w, const float* __restrict__ qn_b,
    const float* __restrict__ kn_w, const float* __restrict__ kn_b,
    _Float16* __restrict__ q_h, _Float16* __restrict__ k_h, _Float16* __restrict__ v_t)
{
  const int blk = blockIdx.x;
  const int w = blk / NPOS, pos = blk % NPOS;
  const int t = threadIdx.x;

  if (pos >= 196) { // zero padding rows/cols
    for (int c = t; c < CH; c += 256) {
      int head = c >> 6, d = c & 63;
      size_t qi = ((size_t)(w * HEADS + head) * NPOS + pos) * DHEAD + d;
      q_h[qi] = (_Float16)0.f;
      k_h[qi] = (_Float16)0.f;
      size_t vi = ((size_t)(w * HEADS + head) * DHEAD + d) * NPOS + pos;
      v_t[vi] = (_Float16)0.f;
    }
    return;
  }

  const _Float16* row = qkv_raw + (size_t)(w * 196 + pos) * 3072;
  __shared__ float red[16];
  __shared__ float stats[2];

#pragma unroll
  for (int which = 0; which < 2; ++which) {
    const _Float16* seg = row + which * CH;
    float vals[4];
    float s = 0.f, ss = 0.f;
#pragma unroll
    for (int j = 0; j < 4; ++j) {
      float v = (float)seg[t + j * 256];
      vals[j] = v; s += v; ss += v * v;
    }
#pragma unroll
    for (int o = 16; o > 0; o >>= 1) {
      s  += __shfl_xor(s, o, 32);
      ss += __shfl_xor(ss, o, 32);
    }
    int wv = t >> 5;
    if ((t & 31) == 0) { red[wv] = s; red[8 + wv] = ss; }
    __syncthreads();
    if (t == 0) {
      float S = 0.f, SS = 0.f;
#pragma unroll
      for (int i = 0; i < 8; ++i) { S += red[i]; SS += red[8 + i]; }
      float mean = S * (1.f / 1024.f);
      float var  = SS * (1.f / 1024.f) - mean * mean;
      stats[0] = mean;
      stats[1] = rsqrtf(var + 1e-5f);
    }
    __syncthreads();
    float mean = stats[0], inv = stats[1];
    const float* gw = which ? kn_w : qn_w;
    const float* gb = which ? kn_b : qn_b;
    float scale = which ? 1.f : 0.125f;      // fold D^-0.5 into q
    _Float16* dst = which ? k_h : q_h;
#pragma unroll
    for (int j = 0; j < 4; ++j) {
      int c = t + j * 256;
      float v = ((vals[j] - mean) * inv * gw[c] + gb[c]) * scale;
      int head = c >> 6, d = c & 63;
      dst[((size_t)(w * HEADS + head) * NPOS + pos) * DHEAD + d] = (_Float16)v;
    }
    __syncthreads();
  }
  // v: copy transposed
#pragma unroll
  for (int j = 0; j < 4; ++j) {
    int c = t + j * 256;
    float v = (float)row[2048 + c];
    int head = c >> 6, d = c & 63;
    v_t[((size_t)(w * HEADS + head) * DHEAD + d) * NPOS + pos] = (_Float16)v;
  }
}

// ---------------- Attention: one wave per (window, head, q-tile of 16 rows) ----------------
__global__ __launch_bounds__(128) void attn_kernel(
    const _Float16* __restrict__ q_h, const _Float16* __restrict__ k_h,
    const _Float16* __restrict__ v_t, _Float16* __restrict__ attn_out)
{
  __shared__ alignas(16) _Float16 pbuf[4][16 * 232];
  const int lane = threadIdx.x & 31;
  const int wv = threadIdx.x >> 5;
  const int n_lane = lane & 15, hi = lane >> 4;

  const int id = blockIdx.x * 4 + wv;         // 0..41599
  const int w = id / (HEADS * 13);
  const int rem = id % (HEADS * 13);
  const int head = rem / 13;
  const int qt = rem % 13;                    // q tiles cover rows 0..207

  const _Float16* qb = q_h + (size_t)(w * HEADS + head) * NPOS * DHEAD;  // [224][64]
  const _Float16* kb = k_h + (size_t)(w * HEADS + head) * NPOS * DHEAD;  // [224][64]
  const _Float16* vb = v_t + (size_t)(w * HEADS + head) * DHEAD * NPOS;  // [64][224]

  v16h aq0 = ld_a(qb + (qt * 16) * DHEAD, DHEAD);
  v16h aq1 = ld_a(qb + (qt * 16) * DHEAD + 32, DHEAD);

  // scores: 16 x 224 (14 tiles)
  v8f s[14];
#pragma unroll
  for (int nt = 0; nt < 14; ++nt) {
    v16h bk0 = ld_b(kb + (nt * 16) * DHEAD, DHEAD);
    v16h bk1 = ld_b(kb + (nt * 16) * DHEAD + 32, DHEAD);
    v8f acc = zero8();
    acc = wmma_f16(aq0, bk0, acc);
    acc = wmma_f16(aq1, bk1, acc);
    s[nt] = acc;
  }

  // softmax per row (row M = r + 8*hi; N spread over 16 lanes of each half)
  _Float16* pl = pbuf[wv];
#pragma unroll
  for (int r = 0; r < 8; ++r) {
    float mx = -3.0e38f;
#pragma unroll
    for (int nt = 0; nt < 14; ++nt) {
      float x = s[nt][r];
      if (nt * 16 + n_lane >= 196) x = -3.0e38f;   // key mask
      mx = fmaxf(mx, x);
    }
#pragma unroll
    for (int o = 1; o < 16; o <<= 1) mx = fmaxf(mx, __shfl_xor(mx, o, 16));
    float sum = 0.f;
#pragma unroll
    for (int nt = 0; nt < 14; ++nt) {
      float e = (nt * 16 + n_lane < 196) ? __expf(s[nt][r] - mx) : 0.f;
      s[nt][r] = e;
      sum += e;
    }
#pragma unroll
    for (int o = 1; o < 16; o <<= 1) sum += __shfl_xor(sum, o, 16);
    float inv = 1.f / sum;
#pragma unroll
    for (int nt = 0; nt < 14; ++nt) {
      pl[(r + 8 * hi) * 232 + nt * 16 + n_lane] = (_Float16)(s[nt][r] * inv);
    }
  }
  __syncthreads();

  // out(16x64) = P(16x224) @ V(224x64)
  v8f o[4];
#pragma unroll
  for (int nd = 0; nd < 4; ++nd) o[nd] = zero8();
#pragma unroll
  for (int kc = 0; kc < 7; ++kc) {
    v16h ap = ld_a(pl + kc * 32, 232);
#pragma unroll
    for (int nd = 0; nd < 4; ++nd) {
      v16h bv = ld_b(vb + (nd * 16) * NPOS + kc * 32, NPOS);
      o[nd] = wmma_f16(ap, bv, o[nd]);
    }
  }

  // scatter valid rows back to gathered (B*4096, 1024) layout
#pragma unroll
  for (int r = 0; r < 8; ++r) {
    int pos = qt * 16 + r + 8 * hi;
    if (pos < 196) {
      int src = window_row_to_src(w * 196 + pos);
      if (src >= 0) {
#pragma unroll
        for (int nd = 0; nd < 4; ++nd) {
          attn_out[(size_t)src * CH + head * DHEAD + nd * 16 + n_lane] = (_Float16)o[nd][r];
        }
      }
    }
  }
}

extern "C" void kernel_launch(void* const* d_in, const int* in_sizes, int n_in,
                              void* d_out, int out_size, void* d_ws, size_t ws_size,
                              hipStream_t stream) {
  const float* x      = (const float*)d_in[0];
  const float* qkv_w  = (const float*)d_in[1];
  const float* qkv_b  = (const float*)d_in[2];
  const float* proj_w = (const float*)d_in[3];
  const float* proj_b = (const float*)d_in[4];
  const float* qn_w   = (const float*)d_in[5];
  const float* qn_b   = (const float*)d_in[6];
  const float* kn_w   = (const float*)d_in[7];
  const float* kn_b   = (const float*)d_in[8];
  float* out = (float*)d_out;

  char* ws = (char*)d_ws;
  _Float16* qkvw_t  = (_Float16*)(ws);                                 // [3072][1024] f16
  _Float16* projw_t = (_Float16*)(ws + 6291456);                       // [1024][1024] f16
  _Float16* qkv_raw = (_Float16*)(ws + 8388608);                       // 39200*3072 f16
  _Float16* attn_h  = qkv_raw;                                         // reuse after LN pass
  _Float16* q_h     = (_Float16*)(ws + 8388608 + 240844800);
  _Float16* k_h     = q_h + (size_t)NWIN * HEADS * NPOS * DHEAD;
  _Float16* v_t     = k_h + (size_t)NWIN * HEADS * NPOS * DHEAD;

  // 0) convert + transpose weights fp32 -> f16 [N][K]
  cvt_transpose_f16<<<(1024 * 3072) / 256, 256, 0, stream>>>(qkv_w, qkvw_t, 1024, 3072);
  cvt_transpose_f16<<<(1024 * 1024) / 256, 256, 0, stream>>>(proj_w, projw_t, 1024, 1024);

  // 1) QKV gather-GEMM: (39200 x 1024) @ (1024 x 3072) + b -> f16
  {
    dim3 grid((WROWS + 127) / 128, 3072 / 128);
    gemm_wmma<true, true><<<grid, 256, 0, stream>>>(
        x, (const _Float16*)nullptr, qkvw_t, qkv_b,
        qkv_raw, (float*)nullptr, WROWS, 3072, 1024);
  }

  // 2) LN(q), LN(k) + scatter to per-(window,head) layouts (with padding to 224)
  ln_scatter<<<NWIN * NPOS, 256, 0, stream>>>(qkv_raw, qn_w, qn_b, kn_w, kn_b, q_h, k_h, v_t);

  // 3) attention (200 windows * 16 heads * 13 q-tiles = 41600 waves, 4 waves/block)
  attn_kernel<<<NWIN * HEADS * 13 / 4, 128, 0, stream>>>(q_h, k_h, v_t, attn_h);

  // 4) proj GEMM: (32768 x 1024) @ (1024 x 1024) + b -> fp32 out
  {
    dim3 grid(XROWS / 128, 1024 / 128);
    gemm_wmma<false, false><<<grid, 256, 0, stream>>>(
        (const float*)nullptr, attn_h, projw_t, proj_b,
        (_Float16*)nullptr, out, XROWS, 1024, 1024);
  }
}
